// TransformerCell_21431886807858
// MI455X (gfx1250) — matrix operand order repaired and checked by reading, not compiled, not run on hardware
//
#include <hip/hip_runtime.h>
#include <math.h>

#define DMODEL 256
#define TN 16          // nodes per workgroup
#define LCTX 7         // context length: 4 children + h + e + s
#define KSTEPS 8       // 256 / 32
#define CSTR 264       // f16 ctx row stride in LDS (256 + 8 pad)

typedef __attribute__((ext_vector_type(16))) _Float16 v16h;
typedef __attribute__((ext_vector_type(8)))  float    v8f;

union Frag16 {
    v16h v;
    _Float16 e[16];
    uint4 u4[2];
};

union H8 {
    _Float16 e[8];
    uint4 u;
};

__device__ __forceinline__ v8f wmma_f16(v16h a, v16h b, v8f c) {
    // D = A(16x32 f16) * B(32x16 f16) + C(16x16 f32)
    return __builtin_amdgcn_wmma_f32_16x16x32_f16(
        /*neg_a=*/false, a, /*neg_b=*/false, b,
        /*c_mod=*/(short)0, c, /*reuse_a=*/false, /*reuse_b=*/false);
}

__device__ __forceinline__ const float* ctx_row_ptr(
    const float* c, const float* h, const float* e, const float* s,
    int node, int l) {
    if (l < 4)  return c + ((size_t)node * 4 + (size_t)l) * DMODEL;
    if (l == 4) return h + (size_t)node * DMODEL;
    if (l == 5) return e + (size_t)node * DMODEL;
    return s + (size_t)node * DMODEL;
}

// A fragment (16x32 f16) from an f16 row staged in LDS.
// Lane layout: lanes 0-15 = rows, K in {lh*8 .. lh*8+7, 16+lh*8 .. 16+lh*8+7}.
__device__ __forceinline__ v16h load_a_frag_lds(const _Float16* row, int kBase, int lh) {
    Frag16 f;
    const _Float16* p0 = row + kBase + lh * 8;
    f.u4[0] = *(const uint4*)(p0);        // K chunk 0 (16B)
    f.u4[1] = *(const uint4*)(p0 + 16);   // K chunk 1 (16B)
    return f.v;
}

// B fragment (32x16 f16) from transposed f16 weights Wt[n*256 + k].
// Lanes: col n, K {kBase+lh*8 .. +7, kBase+16+lh*8 .. +7}.
__device__ __forceinline__ v16h load_b_frag(const _Float16* Wt, int col, int kBase, int lh) {
    Frag16 f;
    const _Float16* p = Wt + (size_t)col * DMODEL + kBase + lh * 8;
    f.u4[0] = *(const uint4*)(p);
    f.u4[1] = *(const uint4*)(p + 16);
    return f.v;
}

// ---------------------------------------------------------------------------
// Prep: convert + transpose the four 256x256 f32 weights into f16 Wt[n][k].
// ---------------------------------------------------------------------------
__global__ void convert_weights_kernel(const float* __restrict__ Wq,
                                       const float* __restrict__ Wk,
                                       const float* __restrict__ Wv,
                                       const float* __restrict__ Wo,
                                       _Float16* __restrict__ wsT) {
    int gid = blockIdx.x * blockDim.x + threadIdx.x;   // 0 .. 4*65536-1
    int w   = gid >> 16;
    int idx = gid & 65535;                             // = k*256 + n
    int k   = idx >> 8;
    int n   = idx & 255;
    const float* src = (w == 0) ? Wq : (w == 1) ? Wk : (w == 2) ? Wv : Wo;
    wsT[(size_t)w * 65536 + (size_t)n * 256 + k] = (_Float16)src[idx];
}

// ---------------------------------------------------------------------------
// Fused cell. One workgroup = 16 nodes, 8 waves; wave w owns column tiles
// {2w, 2w+1} (= head w). ctx is staged ONCE to LDS as f16 (rows l-major:
// row = l*16 + node); A-fragments are then pure ds_load_b128, q stays in
// registers, scores reduce in-register, V is attn-weighted in registers.
// ---------------------------------------------------------------------------
__global__ __launch_bounds__(256) void fused_cell_kernel(
    const float* __restrict__ c, const float* __restrict__ h,
    const float* __restrict__ e, const float* __restrict__ s,
    const _Float16* __restrict__ WqT, const float* __restrict__ bq,
    const _Float16* __restrict__ WkT, const float* __restrict__ bk,
    const _Float16* __restrict__ WvT, const float* __restrict__ bv,
    const _Float16* __restrict__ WoT, const float* __restrict__ bo,
    const float* __restrict__ ln_a, const float* __restrict__ ln_b,
    float* __restrict__ out, int N) {

    // ctx (phases Q..V) aliases oS/LN scratch (phases O..LN, after barrier)
    __shared__ union {
        struct {
            _Float16 ctx[LCTX * 16 * CSTR];   // 112 rows x 264 halves = 59136 B
        } a;
        struct {
            float oS[16 * 260];               // relu(o@Wo+bo), f32
            float lnS[16][16][2];             // partial sums for LN
            float lnMI[16][2];                // mean, 1/(std+eps)
        } b;
    } U;
    __shared__ float    pscore[LCTX][16][16]; // [l][node][16-col group]
    __shared__ float    attnS[16][8][8];      // [node][head][l] (l padded)
    __shared__ _Float16 oA[16 * 264];         // attended output, f16 A rows

    const int node0 = blockIdx.x * TN;
    if (node0 >= N) return;
    const int tid  = threadIdx.x;
    const int wave = tid >> 5;
    const int lane = tid & 31;
    const int lr   = lane & 15;   // row / column index within tile
    const int lh   = lane >> 4;   // lane half
    const int col0 = wave * 32 + lr;              // column for nt = 2*wave
    const int col1 = col0 + 16;                   // column for nt = 2*wave+1

    // ------------- Stage ctx once: f32 global -> f16 LDS ------------------
    // 112 rows x 32 8-float segments; consecutive threads take consecutive
    // segments (coalesced 1KB bursts per row).
    for (int i = tid; i < LCTX * 16 * 32; i += 256) {
        const int row = i >> 5;           // 0..111  (= l*16 + node)
        const int seg = i & 31;           // 8-float segment within the row
        const int l   = row >> 4;
        const int nd  = row & 15;
        const float* src = ctx_row_ptr(c, h, e, s, min(node0 + nd, N - 1), l) + seg * 8;
        float4 a0 = *(const float4*)(src);
        float4 a1 = *(const float4*)(src + 4);
        H8 p;
        p.e[0] = (_Float16)a0.x; p.e[1] = (_Float16)a0.y;
        p.e[2] = (_Float16)a0.z; p.e[3] = (_Float16)a0.w;
        p.e[4] = (_Float16)a1.x; p.e[5] = (_Float16)a1.y;
        p.e[6] = (_Float16)a1.z; p.e[7] = (_Float16)a1.w;
        *(uint4*)&U.a.ctx[row * CSTR + seg * 8] = p.u;
    }
    __syncthreads();

    // ---------------- Q projection: qreg = h @ Wq + bq (registers) --------
    v8f qreg[2];
    {
        v16h A[KSTEPS];
        const _Float16* rowp = U.a.ctx + (4 * 16 + lr) * CSTR;   // h row (l=4)
        #pragma unroll
        for (int ks = 0; ks < KSTEPS; ++ks) A[ks] = load_a_frag_lds(rowp, ks * 32, lh);
        #pragma unroll
        for (int t = 0; t < 2; ++t) {
            const int myCol = t ? col1 : col0;
            v16h B[KSTEPS];
            #pragma unroll
            for (int ks = 0; ks < KSTEPS; ++ks) B[ks] = load_b_frag(WqT, myCol, ks * 32, lh);
            v8f acc = {};
            #pragma unroll
            for (int ks = 0; ks < KSTEPS; ++ks) acc = wmma_f16(A[ks], B[ks], acc);
            const float bias = bq[myCol];
            #pragma unroll
            for (int vi = 0; vi < 8; ++vi) qreg[t][vi] = acc[vi] + bias;
        }
    }

    // -------- K projection; scores formed in-register vs qreg -------------
    {
        const float bk0 = bk[col0];
        const float bk1 = bk[col1];
        for (int l = 0; l < LCTX; ++l) {
            v16h A[KSTEPS];
            const _Float16* rowp = U.a.ctx + (l * 16 + lr) * CSTR;
            #pragma unroll
            for (int ks = 0; ks < KSTEPS; ++ks) A[ks] = load_a_frag_lds(rowp, ks * 32, lh);
            #pragma unroll
            for (int t = 0; t < 2; ++t) {
                const int myCol = t ? col1 : col0;
                const float bias = t ? bk1 : bk0;
                v16h B[KSTEPS];
                #pragma unroll
                for (int ks = 0; ks < KSTEPS; ++ks) B[ks] = load_b_frag(WkT, myCol, ks * 32, lh);
                v8f acc = {};
                #pragma unroll
                for (int ks = 0; ks < KSTEPS; ++ks) acc = wmma_f16(A[ks], B[ks], acc);
                #pragma unroll
                for (int vi = 0; vi < 8; ++vi) {
                    int m = vi + 8 * lh;
                    float p = qreg[t][vi] * (acc[vi] + bias);
                    p += __shfl_xor(p, 1, 32);
                    p += __shfl_xor(p, 2, 32);
                    p += __shfl_xor(p, 4, 32);
                    p += __shfl_xor(p, 8, 32);
                    if (lr == 0) pscore[l][m][wave * 2 + t] = p;
                }
            }
        }
    }
    __syncthreads();

    // ---------------- softmax over L=7 (per node, per head) ---------------
    if (tid < 128) {
        const int m  = tid >> 3;
        const int hd = tid & 7;
        const float scale = 0.17677669529663687f;    // 1/sqrt(32)
        float sc[LCTX];
        float mx = -3.4e38f;
        #pragma unroll
        for (int l = 0; l < LCTX; ++l) {
            sc[l] = (pscore[l][m][2 * hd] + pscore[l][m][2 * hd + 1]) * scale;
            mx = fmaxf(mx, sc[l]);
        }
        float sum = 0.f;
        #pragma unroll
        for (int l = 0; l < LCTX; ++l) { sc[l] = __expf(sc[l] - mx); sum += sc[l]; }
        const float inv = 1.f / sum;
        #pragma unroll
        for (int l = 0; l < LCTX; ++l) attnS[m][hd][l] = sc[l] * inv;
    }
    __syncthreads();

    // ------ V projection; attn-weighted accumulation in registers ---------
    {
        float oacc0[8] = {0.f, 0.f, 0.f, 0.f, 0.f, 0.f, 0.f, 0.f};
        float oacc1[8] = {0.f, 0.f, 0.f, 0.f, 0.f, 0.f, 0.f, 0.f};
        const int hd = wave;                        // head of this wave
        for (int l = 0; l < LCTX; ++l) {
            v16h A[KSTEPS];
            const _Float16* rowp = U.a.ctx + (l * 16 + lr) * CSTR;
            #pragma unroll
            for (int ks = 0; ks < KSTEPS; ++ks) A[ks] = load_a_frag_lds(rowp, ks * 32, lh);
            #pragma unroll
            for (int t = 0; t < 2; ++t) {
                const int myCol = t ? col1 : col0;
                v16h B[KSTEPS];
                #pragma unroll
                for (int ks = 0; ks < KSTEPS; ++ks) B[ks] = load_b_frag(WvT, myCol, ks * 32, lh);
                v8f acc = {};
                #pragma unroll
                for (int ks = 0; ks < KSTEPS; ++ks) acc = wmma_f16(A[ks], B[ks], acc);
                float* oacc = t ? oacc1 : oacc0;
                #pragma unroll
                for (int vi = 0; vi < 8; ++vi) {
                    int m = vi + 8 * lh;
                    oacc[vi] += attnS[m][hd][l] * acc[vi];
                }
            }
        }
        const float bv0 = bv[col0];                 // sum(attn)=1 -> +bv once
        const float bv1 = bv[col1];
        #pragma unroll
        for (int vi = 0; vi < 8; ++vi) {
            int m = vi + 8 * lh;
            oA[m * 264 + col0] = (_Float16)(oacc0[vi] + bv0);
            oA[m * 264 + col1] = (_Float16)(oacc1[vi] + bv1);
        }
    }
    __syncthreads();   // ctx dead from here; U.b (oS/LN) becomes live

    // ---------------- O projection + bias + ReLU -> oS ---------------------
    {
        v16h A[KSTEPS];
        const _Float16* rowp = oA + (size_t)lr * 264;
        #pragma unroll
        for (int ks = 0; ks < KSTEPS; ++ks) A[ks] = load_a_frag_lds(rowp, ks * 32, lh);
        #pragma unroll
        for (int t = 0; t < 2; ++t) {
            const int myCol = t ? col1 : col0;
            v16h B[KSTEPS];
            #pragma unroll
            for (int ks = 0; ks < KSTEPS; ++ks) B[ks] = load_b_frag(WoT, myCol, ks * 32, lh);
            v8f acc = {};
            #pragma unroll
            for (int ks = 0; ks < KSTEPS; ++ks) acc = wmma_f16(A[ks], B[ks], acc);
            const float bias = bo[myCol];
            #pragma unroll
            for (int vi = 0; vi < 8; ++vi) {
                int m = vi + 8 * lh;
                float v = acc[vi] + bias;
                U.b.oS[m * 260 + myCol] = v > 0.f ? v : 0.f;
            }
        }
    }
    __syncthreads();

    // ---------------- LayerNorm (unbiased var, eps added to std) ----------
    {
        const int m = tid >> 4;
        const int j = tid & 15;
        float sum = 0.f, sq = 0.f;
        #pragma unroll
        for (int i = 0; i < 16; ++i) {
            float v = U.b.oS[m * 260 + j * 16 + i];
            sum += v; sq += v * v;
        }
        U.b.lnS[m][j][0] = sum;
        U.b.lnS[m][j][1] = sq;
    }
    __syncthreads();
    if (tid < 16) {
        const int m = tid;
        float sum = 0.f, sq = 0.f;
        #pragma unroll
        for (int j = 0; j < 16; ++j) { sum += U.b.lnS[m][j][0]; sq += U.b.lnS[m][j][1]; }
        const float mean = sum * (1.f / 256.f);
        float var = (sq - 256.f * mean * mean) * (1.f / 255.f);
        var = var < 0.f ? 0.f : var;
        U.b.lnMI[m][0] = mean;
        U.b.lnMI[m][1] = 1.f / (sqrtf(var) + 1e-6f);
    }
    __syncthreads();
    {
        const int m = tid >> 4;
        const int j = tid & 15;
        if (node0 + m < N) {
            const float mean = U.b.lnMI[m][0];
            const float inv  = U.b.lnMI[m][1];
            float* op = out + (size_t)(node0 + m) * DMODEL + j * 16;
            #pragma unroll
            for (int i4 = 0; i4 < 4; ++i4) {
                const int col = j * 16 + i4 * 4;
                float4 r;
                r.x = ln_a[col + 0] * (U.b.oS[m * 260 + col + 0] - mean) * inv + ln_b[col + 0];
                r.y = ln_a[col + 1] * (U.b.oS[m * 260 + col + 1] - mean) * inv + ln_b[col + 1];
                r.z = ln_a[col + 2] * (U.b.oS[m * 260 + col + 2] - mean) * inv + ln_b[col + 2];
                r.w = ln_a[col + 3] * (U.b.oS[m * 260 + col + 3] - mean) * inv + ln_b[col + 3];
                *(float4*)(op + i4 * 4) = r;
            }
        }
    }
}

extern "C" void kernel_launch(void* const* d_in, const int* in_sizes, int n_in,
                              void* d_out, int out_size, void* d_ws, size_t ws_size,
                              hipStream_t stream) {
    const float* c    = (const float*)d_in[0];
    const float* h    = (const float*)d_in[1];
    const float* e    = (const float*)d_in[2];
    const float* s    = (const float*)d_in[3];
    const float* Wq   = (const float*)d_in[4];
    const float* bq   = (const float*)d_in[5];
    const float* Wk   = (const float*)d_in[6];
    const float* bk   = (const float*)d_in[7];
    const float* Wv   = (const float*)d_in[8];
    const float* bv   = (const float*)d_in[9];
    const float* Wo   = (const float*)d_in[10];
    const float* bo   = (const float*)d_in[11];
    const float* ln_a = (const float*)d_in[12];
    const float* ln_b = (const float*)d_in[13];
    float* out = (float*)d_out;
    const int N = in_sizes[1] / DMODEL;

    _Float16* wsT = (_Float16*)d_ws;   // 4 x 256x256 f16 transposed weights

    convert_weights_kernel<<<(4 * 65536) / 256, 256, 0, stream>>>(Wq, Wk, Wv, Wo, wsT);

    const int nBlocks = (N + TN - 1) / TN;
    fused_cell_kernel<<<nBlocks, 256, 0, stream>>>(
        c, h, e, s,
        wsT + 0 * 65536, bq,
        wsT + 1 * 65536, bk,
        wsT + 2 * 65536, bv,
        wsT + 3 * 65536, bo,
        ln_a, ln_b, out, N);
}
